// GPTLanguageModel_40381282517702
// MI455X (gfx1250) — matrix operand order repaired and checked by reading, not compiled, not run on hardware
//
#include <hip/hip_runtime.h>

// ---------------------------------------------------------------------------
// GPT forward for MI455X (gfx1250, wave32, WMMA).
// All GEMMs (qkv, attention scores, P*V, proj, fc1, fc2, head) run through
// v_wmma_f32_16x16x32_bf16. Global->LDS staging uses CDNA5 async-LDS loads
// (global_load_async_to_lds_b128, ASYNCcnt) with double buffering.
// fp32 weights are converted once per call into bf16 copies in d_ws; the
// residual stream stays fp32. Workspace usage ~351 MB.
// ---------------------------------------------------------------------------

typedef __bf16 bf16_t;
typedef __attribute__((ext_vector_type(16))) __bf16 v16bf;
typedef __attribute__((ext_vector_type(8)))  float  v8f;
typedef __attribute__((ext_vector_type(4)))  unsigned int u32x4;

struct frag_bits { u32x4 lo, hi; };

__device__ __forceinline__ v16bf make_frag(u32x4 lo, u32x4 hi) {
  frag_bits t;
  t.lo = lo; t.hi = hi;
  return __builtin_bit_cast(v16bf, t);
}

// A-matrix fragment, 16x32 bf16, source row-major [16][ld].
// Per ISA 7.12.2: lanes 0-15 (M=lane) hold K={0..7,16..23}; lanes 16-31 hold
// K={8..15,24..31}. Packed-pair (even k) dword groups -> two 16B loads.
__device__ __forceinline__ v16bf load_a_frag(const bf16_t* p, int ld) {
  int lane = threadIdx.x & 31;
  int row  = lane & 15;
  int hi   = lane >> 4;
  const bf16_t* r = p + row * ld + hi * 8;
  u32x4 lo = *(const u32x4*)(r);
  u32x4 hx = *(const u32x4*)(r + 16);
  return make_frag(lo, hx);
}

// B-matrix fragment, 32x16 bf16 (KxN), sourced from 16 rows of k-pairs:
// element (k,n) = src[n*ld + k]. lanes 0-15 (N=lane) hold K={0..7,16..23},
// lanes 16-31 hold K={8..15,24..31} (same order as the A fragment).
__device__ __forceinline__ v16bf load_b_frag(const bf16_t* p, int ld) {
  int lane = threadIdx.x & 31;
  int row  = lane & 15;
  int hi   = lane >> 4;
  const bf16_t* r = p + row * ld + hi * 8;
  u32x4 lo = *(const u32x4*)(r);
  u32x4 hx = *(const u32x4*)(r + 16);
  return make_frag(lo, hx);
}

__device__ __forceinline__ v8f wmma_bf16(v16bf a, v16bf b, v8f c) {
  return __builtin_amdgcn_wmma_f32_16x16x32_bf16(
      /*neg_a=*/false, a, /*neg_b=*/false, b,
      /*c_mod=*/(short)0, c, /*reuse_a=*/false, /*reuse_b=*/false);
}

// CDNA5 async global->LDS copy, 16B per lane, GV addressing (64-bit vaddr).
// LDS destination = low 32 bits of the generic pointer (flat->LDS truncation
// per ISA 10.2 aperture rules). Tracked with ASYNCcnt.
__device__ __forceinline__ void async_load_b128(unsigned lds_off, const void* g) {
  asm volatile("global_load_async_to_lds_b128 %0, %1, off"
               :: "v"(lds_off), "v"((unsigned long long)(size_t)g)
               : "memory");
}
__device__ __forceinline__ void wait_async0() {
  asm volatile("s_wait_asynccnt 0x0" ::: "memory");
}

// ---------------------------------------------------------------------------
// fp32 -> bf16 conversion (weights)
// ---------------------------------------------------------------------------
__global__ __launch_bounds__(256)
void cvt_bf16_kernel(const float* __restrict__ in, bf16_t* __restrict__ out,
                     unsigned long long n) {
  unsigned long long i = (unsigned long long)blockIdx.x * blockDim.x + threadIdx.x;
  unsigned long long stride = (unsigned long long)gridDim.x * blockDim.x;
  for (; i < n; i += stride) out[i] = (bf16_t)in[i];
}

// ---------------------------------------------------------------------------
// Embedding: x[b,s,:] = tok_emb[idx[b,s],:] + pos_emb[s,:]
// ---------------------------------------------------------------------------
__global__ __launch_bounds__(256)
void embed_kernel(const int* __restrict__ idx, const float* __restrict__ tok,
                  const float* __restrict__ pos, float* __restrict__ x) {
  int row = blockIdx.x;            // b*S + s, S=1024
  int s   = row & 1023;
  int t   = idx[row];
  int tid = threadIdx.x;
#pragma unroll
  for (int j = 0; j < 4; ++j) {
    int c = tid * 4 + j;
    x[(size_t)row * 1024 + c] =
        tok[(size_t)t * 1024 + c] + pos[(size_t)s * 1024 + c];
  }
}

// ---------------------------------------------------------------------------
// LayerNorm (fp32 in, bf16 out), one block per row, D=1024
// ---------------------------------------------------------------------------
__global__ __launch_bounds__(256)
void layernorm_kernel(const float* __restrict__ x, const float* __restrict__ gain,
                      const float* __restrict__ bias, bf16_t* __restrict__ out) {
  __shared__ float red[18];
  int row = blockIdx.x;
  int tid = threadIdx.x;
  const float* xr = x + (size_t)row * 1024;
  float4 v = *(const float4*)&xr[tid * 4];
  float sum = v.x + v.y + v.z + v.w;
  float sq  = v.x * v.x + v.y * v.y + v.z * v.z + v.w * v.w;
#pragma unroll
  for (int m = 16; m >= 1; m >>= 1) {
    sum += __shfl_xor(sum, m, 32);
    sq  += __shfl_xor(sq,  m, 32);
  }
  int wave = tid >> 5, lane = tid & 31;
  if (lane == 0) { red[wave] = sum; red[8 + wave] = sq; }
  __syncthreads();
  if (tid == 0) {
    float a = 0.f, b = 0.f;
    for (int i = 0; i < 8; ++i) { a += red[i]; b += red[8 + i]; }
    float mu  = a * (1.f / 1024.f);
    float var = b * (1.f / 1024.f) - mu * mu;
    red[16] = mu;
    red[17] = rsqrtf(var + 1e-5f);
  }
  __syncthreads();
  float mu = red[16], rstd = red[17];
#pragma unroll
  for (int j = 0; j < 4; ++j) {
    int c = tid * 4 + j;
    float val = ((&v.x)[j] - mu) * rstd * gain[c] + bias[c];
    out[(size_t)row * 1024 + c] = (bf16_t)val;
  }
}

// ---------------------------------------------------------------------------
// Generic WMMA GEMM: C[M,N] = A[M,K] * W[N,K]^T (+bias) (+epilogue)
// Block tile 128(M) x 128(N), K step 32, double-buffered async-LDS staging.
// 8 waves; each wave computes 32x64 of C (8 accumulators, 8 WMMAs/k-step).
// MODE 0: bf16 out           MODE 1: bf16 out + ReLU
// MODE 2: fp32 residual add  MODE 3: qkv scatter to Q/K/V [B,H,S,64] bf16
// MODE 4: fp32 out with N-bound guard (head, N=50257)
// ---------------------------------------------------------------------------
template <int MODE>
__global__ __launch_bounds__(256)
void gemm_bf16_kernel(const bf16_t* __restrict__ A, const bf16_t* __restrict__ W,
                      const float* __restrict__ bias,
                      void* __restrict__ out0, void* __restrict__ out1,
                      void* __restrict__ out2, float* __restrict__ resid,
                      int N, int K) {
  __shared__ bf16_t sA[2][128 * 32];   // 2 x 8 KB
  __shared__ bf16_t sB[2][128 * 32];   // 2 x 8 KB

  int tid  = threadIdx.x;
  int wave = tid >> 5;
  int lane = tid & 31;
  int mw   = wave >> 1;             // 0..3 -> 32 rows of C
  int nw   = wave & 1;              // 0..1 -> 64 cols of C
  int bm   = blockIdx.y * 128;
  int bn   = blockIdx.x * 128;

  v8f acc[2][4];
#pragma unroll
  for (int mi = 0; mi < 2; ++mi)
#pragma unroll
    for (int nt = 0; nt < 4; ++nt)
#pragma unroll
      for (int r = 0; r < 8; ++r) acc[mi][nt][r] = 0.f;

  // Staging coordinates: each thread copies 16 bf16 (2 async b128) of A and W.
  int off16 = tid * 16;             // 0..4095
  int srow  = off16 >> 5;           // 0..127
  int scol  = off16 & 31;           // 0 or 16
  const bf16_t* agbase = &A[(size_t)(bm + srow) * K + scol];
  int wr = bn + srow; if (wr >= N) wr = N - 1;   // clamp ragged N (head)
  const bf16_t* wgbase = &W[(size_t)wr * K + scol];

  int nk = K >> 5;
  // Issue tile 0.
  {
    unsigned la = (unsigned)(size_t)&sA[0][off16];
    unsigned lb = (unsigned)(size_t)&sB[0][off16];
    async_load_b128(la,      agbase);
    async_load_b128(la + 16, agbase + 8);
    async_load_b128(lb,      wgbase);
    async_load_b128(lb + 16, wgbase + 8);
  }

  for (int it = 0; it < nk; ++it) {
    int cur = it & 1;
    wait_async0();        // my tile-it copies have landed in LDS
    __syncthreads();      // everyone's have; nobody still reads buf[cur^1]
    if (it + 1 < nk) {    // prefetch next tile into the other buffer
      int k0 = (it + 1) << 5;
      unsigned la = (unsigned)(size_t)&sA[cur ^ 1][off16];
      unsigned lb = (unsigned)(size_t)&sB[cur ^ 1][off16];
      async_load_b128(la,      agbase + k0);
      async_load_b128(la + 16, agbase + k0 + 8);
      async_load_b128(lb,      wgbase + k0);
      async_load_b128(lb + 16, wgbase + k0 + 8);
    }
    v16bf a0 = load_a_frag(&sA[cur][(mw * 32) * 32], 32);
    v16bf a1 = load_a_frag(&sA[cur][(mw * 32 + 16) * 32], 32);
#pragma unroll
    for (int nt = 0; nt < 4; ++nt) {
      v16bf b = load_b_frag(&sB[cur][(nw * 64 + nt * 16) * 32], 32);
      acc[0][nt] = wmma_bf16(a0, b, acc[0][nt]);
      acc[1][nt] = wmma_bf16(a1, b, acc[1][nt]);
    }
  }

  // Epilogue. C layout: lane holds column n = lane&15, rows r + (lane>=16)*8.
  int cn = lane & 15;
#pragma unroll
  for (int mi = 0; mi < 2; ++mi) {
    int rbase = bm + mw * 32 + mi * 16 + ((lane >> 4) << 3);
#pragma unroll
    for (int nt = 0; nt < 4; ++nt) {
      int gn = bn + nw * 64 + nt * 16 + cn;
      float bv = 0.f;
      if (MODE != 3) bv = (gn < N) ? bias[gn] : 0.f;
#pragma unroll
      for (int r = 0; r < 8; ++r) {
        int gm = rbase + r;
        float v = acc[mi][nt][r] + bv;
        if (MODE == 0) {
          ((bf16_t*)out0)[(size_t)gm * N + gn] = (bf16_t)v;
        } else if (MODE == 1) {
          ((bf16_t*)out0)[(size_t)gm * N + gn] = (bf16_t)fmaxf(v, 0.f);
        } else if (MODE == 2) {
          resid[(size_t)gm * N + gn] += v;
        } else if (MODE == 4) {
          if (gn < N)
            ((float*)out0)[(size_t)gm * (size_t)N + gn] = v;
        } else {  // MODE == 3: per-head interleaved qkv scatter (3*HD = 192)
          int h = gn / 192, rr = gn - h * 192;
          int b = gm >> 10, s = gm & 1023;
          size_t di = (((size_t)(b * 16 + h) * 1024) + s) * 64 + (rr & 63);
          bf16_t bw = (bf16_t)v;
          if (rr < 64)        ((bf16_t*)out0)[di] = bw;   // Q
          else if (rr < 128)  ((bf16_t*)out1)[di] = bw;   // K
          else                ((bf16_t*)out2)[di] = bw;   // V
        }
      }
    }
  }
}

// ---------------------------------------------------------------------------
// Flash-style causal attention. Q/K/V bf16 [B,H,S,64]. Block = 4 waves; each
// wave owns a 16-row q-tile; K-chunk (32x64) and transposed V-chunk shared.
// Q/K staged with async-LDS loads. Out: attn bf16 [B,S,D] (d = h*64 + col).
// ---------------------------------------------------------------------------
__global__ __launch_bounds__(128)
void attn_kernel(const bf16_t* __restrict__ Q, const bf16_t* __restrict__ Km,
                 const bf16_t* __restrict__ V, bf16_t* __restrict__ out) {
  __shared__ bf16_t sQ[4][16 * 64];   // 8 KB
  __shared__ bf16_t sK[32 * 64];      // 4 KB
  __shared__ bf16_t sVt[64 * 32];     // 4 KB (V^T: [d][key])
  __shared__ bf16_t sP[4][16 * 32];   // 4 KB

  int tid  = threadIdx.x;
  int wave = tid >> 5;
  int lane = tid & 31;
  int bh = blockIdx.y;               // b*H + h
  int b  = bh >> 4;
  int h  = bh & 15;
  int qt = blockIdx.x * 4 + wave;    // q tile (16 rows)
  const size_t base = (size_t)bh * 1024 * 64;

  // Stage this wave's Q tile (16x64) via async-LDS.
  {
    int row = lane >> 1;
    int col = (lane & 1) * 32;
    const bf16_t* src = Q + base + (size_t)(qt * 16 + row) * 64 + col;
    unsigned dst = (unsigned)(size_t)&sQ[wave][row * 64 + col];
    async_load_b128(dst,      src);
    async_load_b128(dst + 16, src + 8);
    async_load_b128(dst + 32, src + 16);
    async_load_b128(dst + 48, src + 24);
  }
  wait_async0();
  __syncthreads();
  v16bf aq0 = load_a_frag(&sQ[wave][0], 64);    // Q cols 0..31
  v16bf aq1 = load_a_frag(&sQ[wave][32], 64);   // Q cols 32..63

  float rm[8], rs[8];
  v8f acc[4];
#pragma unroll
  for (int r = 0; r < 8; ++r) { rm[r] = -1e30f; rs[r] = 0.f; }
#pragma unroll
  for (int nt = 0; nt < 4; ++nt)
#pragma unroll
    for (int r = 0; r < 8; ++r) acc[nt][r] = 0.f;

  int cn = lane & 15;
  int rb = qt * 16 + ((lane >> 4) << 3);
  int nchunks = blockIdx.x * 2 + 2;   // covers causal range of all 4 q-tiles

  for (int kc = 0; kc < nchunks; ++kc) {
    __syncthreads();   // previous chunk fully consumed
    // Stage K chunk async: rows kc*32..+31, 64 cols; 128 threads x 16 bf16.
    {
      int off = tid * 16;
      int row = off >> 6;
      int col = off & 63;
      const bf16_t* src = Km + base + (size_t)(kc * 32 + row) * 64 + col;
      unsigned dst = (unsigned)(size_t)&sK[off];
      async_load_b128(dst,      src);
      async_load_b128(dst + 16, src + 8);
    }
    // Stage V^T chunk (transpose scatter): sVt[d*32 + key] = V[key][d].
    {
      int key = tid & 31;
      int dg  = tid >> 5;             // 0..3 -> d group of 16
      const bf16_t* src = V + base + (size_t)(kc * 32 + key) * 64 + dg * 16;
      bf16_t tmp[16];
      *(u32x4*)&tmp[0] = *(const u32x4*)(src);
      *(u32x4*)&tmp[8] = *(const u32x4*)(src + 8);
#pragma unroll
      for (int j = 0; j < 16; ++j) sVt[(dg * 16 + j) * 32 + key] = tmp[j];
    }
    wait_async0();
    __syncthreads();

    // Scores: S = Q (16x64) x K_chunk^T (64x32) -> two 16x16 tiles, 4 WMMAs.
    v8f sc[2];
#pragma unroll
    for (int nt = 0; nt < 2; ++nt) {
      v8f s;
#pragma unroll
      for (int r = 0; r < 8; ++r) s[r] = 0.f;
      s = wmma_bf16(aq0, load_b_frag(&sK[nt * 16 * 64], 64), s);
      s = wmma_bf16(aq1, load_b_frag(&sK[nt * 16 * 64 + 32], 64), s);
      sc[nt] = s;
    }

    // Scale + causal mask + online softmax (row reductions over 16 lanes).
    float mnew[8];
#pragma unroll
    for (int r = 0; r < 8; ++r) {
      int grow = rb + r;
      float v0 = sc[0][r] * 0.125f;   // 1/sqrt(64)
      float v1 = sc[1][r] * 0.125f;
      if (kc * 32 + cn      > grow) v0 = -1e30f;
      if (kc * 32 + 16 + cn > grow) v1 = -1e30f;
      sc[0][r] = v0; sc[1][r] = v1;
      float m = fmaxf(v0, v1);
      m = fmaxf(m, __shfl_xor(m, 1, 32));
      m = fmaxf(m, __shfl_xor(m, 2, 32));
      m = fmaxf(m, __shfl_xor(m, 4, 32));
      m = fmaxf(m, __shfl_xor(m, 8, 32));
      mnew[r] = fmaxf(rm[r], m);
    }
#pragma unroll
    for (int r = 0; r < 8; ++r) {
      float f = __expf(rm[r] - mnew[r]);
      rm[r] = mnew[r];
      float p0 = __expf(sc[0][r] - rm[r]);
      float p1 = __expf(sc[1][r] - rm[r]);
      float ps = p0 + p1;
      ps += __shfl_xor(ps, 1, 32);
      ps += __shfl_xor(ps, 2, 32);
      ps += __shfl_xor(ps, 4, 32);
      ps += __shfl_xor(ps, 8, 32);
      rs[r] = rs[r] * f + ps;
#pragma unroll
      for (int nt = 0; nt < 4; ++nt) acc[nt][r] *= f;
      int prow = r + ((lane >> 4) << 3);
      sP[wave][prow * 32 + cn]      = (bf16_t)p0;   // C-layout -> LDS row-major
      sP[wave][prow * 32 + 16 + cn] = (bf16_t)p1;
    }
    // Same-wave LDS is in-order (DScnt): P store -> A-frag load is safe.
    v16bf ap = load_a_frag(&sP[wave][0], 32);
#pragma unroll
    for (int nt = 0; nt < 4; ++nt)
      acc[nt] = wmma_bf16(ap, load_b_frag(&sVt[nt * 16 * 32], 32), acc[nt]);
  }

  // Normalize and write out (bf16 [B,S,D]).
#pragma unroll
  for (int nt = 0; nt < 4; ++nt) {
#pragma unroll
    for (int r = 0; r < 8; ++r) {
      int srow = qt * 16 + r + ((lane >> 4) << 3);
      float o = acc[nt][r] / rs[r];
      out[(size_t)(b * 1024 + srow) * 1024 + h * 64 + nt * 16 + cn] = (bf16_t)o;
    }
  }
}

// ---------------------------------------------------------------------------
// Host orchestration
// ---------------------------------------------------------------------------
extern "C" void kernel_launch(void* const* d_in, const int* in_sizes, int n_in,
                              void* d_out, int out_size, void* d_ws, size_t ws_size,
                              hipStream_t stream) {
  (void)in_sizes; (void)n_in; (void)out_size; (void)ws_size;

  const int*   idx   = (const int*)  d_in[0];
  const float* tok   = (const float*)d_in[1];
  const float* pos   = (const float*)d_in[2];
  const float* qkvw  = (const float*)d_in[3];
  const float* projw = (const float*)d_in[4];
  const float* projb = (const float*)d_in[5];
  const float* ln1s  = (const float*)d_in[6];
  const float* ln1b  = (const float*)d_in[7];
  const float* fc1w  = (const float*)d_in[8];
  const float* fc1b  = (const float*)d_in[9];
  const float* fc2w  = (const float*)d_in[10];
  const float* fc2b  = (const float*)d_in[11];
  const float* ln2s  = (const float*)d_in[12];
  const float* ln2b  = (const float*)d_in[13];
  const float* lnfs  = (const float*)d_in[14];
  const float* lnfb  = (const float*)d_in[15];
  const float* headw = (const float*)d_in[16];
  const float* headb = (const float*)d_in[17];
  float* logits = (float*)d_out;

  const size_t Dm = 1024, FF = 4096, Vv = 50257, Lc = 8;
  const size_t Msz = 2 * 1024;            // B*S
  const size_t nQkvW = Lc * 3 * Dm * Dm;  // 25,165,824
  const size_t nProjW = Lc * Dm * Dm;     //  8,388,608
  const size_t nFc1W = Lc * FF * Dm;      // 33,554,432
  const size_t nFc2W = Lc * Dm * FF;      // 33,554,432
  const size_t nHeadW = Vv * Dm;          // 51,463,168

  char* ws = (char*)d_ws;
  size_t off = 0;
  auto alloc = [&](size_t bytes) -> char* {
    char* p = ws + off;
    off = (off + bytes + 255) & ~(size_t)255;
    return p;
  };
  bf16_t* qkvw_b  = (bf16_t*)alloc(nQkvW  * 2);
  bf16_t* projw_b = (bf16_t*)alloc(nProjW * 2);
  bf16_t* fc1w_b  = (bf16_t*)alloc(nFc1W  * 2);
  bf16_t* fc2w_b  = (bf16_t*)alloc(nFc2W  * 2);
  bf16_t* headw_b = (bf16_t*)alloc(nHeadW * 2);
  float*  x       = (float*) alloc(Msz * Dm * 4);   // residual stream fp32
  bf16_t* hbf     = (bf16_t*)alloc(Msz * Dm * 2);   // LN output
  bf16_t* qbf     = (bf16_t*)alloc(Msz * Dm * 2);   // [B,H,S,64]
  bf16_t* kbf     = (bf16_t*)alloc(Msz * Dm * 2);
  bf16_t* vbf     = (bf16_t*)alloc(Msz * Dm * 2);
  bf16_t* attnbf  = (bf16_t*)alloc(Msz * Dm * 2);   // [B,S,D]
  bf16_t* ffbf    = (bf16_t*)alloc(Msz * FF * 2);   // ReLU output

  // Weight conversion fp32 -> bf16 (deterministic, done every call).
  cvt_bf16_kernel<<<4096, 256, 0, stream>>>(qkvw,  qkvw_b,  nQkvW);
  cvt_bf16_kernel<<<4096, 256, 0, stream>>>(projw, projw_b, nProjW);
  cvt_bf16_kernel<<<4096, 256, 0, stream>>>(fc1w,  fc1w_b,  nFc1W);
  cvt_bf16_kernel<<<4096, 256, 0, stream>>>(fc2w,  fc2w_b,  nFc2W);
  cvt_bf16_kernel<<<8192, 256, 0, stream>>>(headw, headw_b, nHeadW);

  // Embedding
  embed_kernel<<<Msz, 256, 0, stream>>>(idx, tok, pos, x);

  for (int l = 0; l < (int)Lc; ++l) {
    const bf16_t* wq = qkvw_b  + (size_t)l * 3 * Dm * Dm;
    const bf16_t* wp = projw_b + (size_t)l * Dm * Dm;
    const bf16_t* w1 = fc1w_b  + (size_t)l * FF * Dm;
    const bf16_t* w2 = fc2w_b  + (size_t)l * Dm * FF;

    // h = LN1(x)
    layernorm_kernel<<<Msz, 256, 0, stream>>>(x, ln1s + l * Dm, ln1b + l * Dm, hbf);
    // qkv = h @ qkv_w^T, scatter to Q/K/V [B,H,S,64]
    gemm_bf16_kernel<3><<<dim3(24, 16), 256, 0, stream>>>(
        hbf, wq, nullptr, qbf, kbf, vbf, nullptr, 3072, 1024);
    // causal attention
    attn_kernel<<<dim3(16, 32), 128, 0, stream>>>(qbf, kbf, vbf, attnbf);
    // x += attn @ proj_w^T + proj_b
    gemm_bf16_kernel<2><<<dim3(8, 16), 256, 0, stream>>>(
        attnbf, wp, projb + l * Dm, nullptr, nullptr, nullptr, x, 1024, 1024);
    // h = LN2(x)
    layernorm_kernel<<<Msz, 256, 0, stream>>>(x, ln2s + l * Dm, ln2b + l * Dm, hbf);
    // ff = relu(h @ fc1_w^T + fc1_b)
    gemm_bf16_kernel<1><<<dim3(32, 16), 256, 0, stream>>>(
        hbf, w1, fc1b + l * FF, ffbf, nullptr, nullptr, nullptr, 4096, 1024);
    // x += ff @ fc2_w^T + fc2_b
    gemm_bf16_kernel<2><<<dim3(8, 16), 256, 0, stream>>>(
        ffbf, w2, fc2b + l * Dm, nullptr, nullptr, nullptr, x, 1024, 4096);
  }

  // Final LN + head projection (ragged N = 50257, fp32 logits)
  layernorm_kernel<<<Msz, 256, 0, stream>>>(x, lnfs, lnfb, hbf);
  gemm_bf16_kernel<4><<<dim3(393, 16), 256, 0, stream>>>(
      hbf, headw_b, headb, logits, nullptr, nullptr, nullptr, 50257, 1024);
}
